// GCLSTMModel_50483045597457
// MI455X (gfx1250) — matrix operand order repaired and checked by reading, not compiled, not run on hardware
//
#include <hip/hip_runtime.h>
#include <hip/hip_bf16.h>

#define N_NODES 50000
#define N_PAD   50016            // multiple of 32
#define E_EDGES 1200000
#define HID     64
#define LAY     2

typedef _Float16 v8h  __attribute__((ext_vector_type(8)));
typedef _Float16 v16h __attribute__((ext_vector_type(16)));
typedef float    v8f  __attribute__((ext_vector_type(8)));

__device__ __forceinline__ float sigf(float x) {
    return 1.0f / (1.0f + __expf(-x));
}

__device__ __forceinline__ void atomic_addf(float* p, float v) {
    __hip_atomic_fetch_add(p, v, __ATOMIC_RELAXED, __HIP_MEMORY_SCOPE_AGENT);
}

// ---------------- utility: zero a float buffer (count divisible by 4) -------
__global__ void zero_f32(float* __restrict__ p, int n4) {
    int i = blockIdx.x * blockDim.x + threadIdx.x;
    if (i < n4) {
        float4 z; z.x = 0.f; z.y = 0.f; z.z = 0.f; z.w = 0.f;
        ((float4*)p)[i] = z;
    }
}

// ---------------- degree accumulation ---------------------------------------
__global__ void deg_kernel(const int* __restrict__ row, const float* __restrict__ ew,
                           float* __restrict__ deg) {
    int e = blockIdx.x * blockDim.x + threadIdx.x;
    if (e < E_EDGES) atomic_addf(&deg[row[e]], ew[e]);
}

// ---------------- normalized edge weights: nw = -dinv[row]*w*dinv[col] ------
__global__ void nw_kernel(const int* __restrict__ row, const int* __restrict__ col,
                          const float* __restrict__ ew, const float* __restrict__ deg,
                          float* __restrict__ nw) {
    int e = blockIdx.x * blockDim.x + threadIdx.x;
    if (e >= E_EDGES) return;
    float dr = deg[row[e]], dc = deg[col[e]];
    float ir = dr > 0.f ? rsqrtf(dr) : 0.f;
    float ic = dc > 0.f ? rsqrtf(dc) : 0.f;
    nw[e] = -ir * ew[e] * ic;
}

// ---------------- sparse propagation: dst[row] += nw * src[col] -------------
// 16 threads per edge, 4 features each (float4 gather, 4 scatter atomics)
__global__ void prop_kernel(const int* __restrict__ row, const int* __restrict__ col,
                            const float* __restrict__ nw, const float* __restrict__ src,
                            float* __restrict__ dst) {
    int i = blockIdx.x * blockDim.x + threadIdx.x;
    if (i >= E_EDGES * 16) return;
    int e = i >> 4;
    int q = (i & 15) * 4;
    float w = nw[e];
    const float4 v = *(const float4*)(src + (size_t)col[e] * HID + q);
    float* o = dst + (size_t)row[e] * HID + q;
    atomic_addf(o + 0, w * v.x);
    atomic_addf(o + 1, w * v.y);
    atomic_addf(o + 2, w * v.z);
    atomic_addf(o + 3, w * v.w);
}

// ---------------- fused weight repack: Wt[c][k] (f16, K-contiguous) ---------
__global__ void wprep_kernel(const float* __restrict__ Wx, const float* __restrict__ Wc,
                             const float* __restrict__ bc, const float* __restrict__ bg,
                             int l, _Float16* __restrict__ Wt, float* __restrict__ bias) {
    int i = blockIdx.x * blockDim.x + threadIdx.x;
    if (i >= 256 * 256) return;
    int k = i & 255, cidx = i >> 8;
    int g = cidx >> 6, o = cidx & 63;
    float v;
    if (k < 64) {
        v = Wx[(((size_t)l * 4 + g) * 64 + k) * 64 + o];
    } else {
        int kk = (k - 64) >> 6, r = k & 63;
        v = Wc[((((size_t)l * 4 + g) * 3 + kk) * 64 + r) * 64 + o];
    }
    Wt[(size_t)cidx * 256 + k] = (_Float16)v;
    if (k == 0)
        bias[cidx] = bc[((size_t)l * 4 + g) * 64 + o] + bg[((size_t)l * 4 + g) * 64 + o];
}

// ---------------- pack X = [inp | h | T1 | 2*T2p - h] into f16 -------------
__global__ void xpack_kernel(const float* __restrict__ inp, const float* __restrict__ hl,
                             const float* __restrict__ p1, const float* __restrict__ p2,
                             _Float16* __restrict__ Xh) {
    int i = blockIdx.x * blockDim.x + threadIdx.x;
    if (i >= N_PAD * 256) return;
    int n = i >> 8, k = i & 255;
    int ns = n < N_NODES ? n : N_NODES - 1;   // clamp padded rows
    float v;
    if (k < 64)       v = inp[(size_t)ns * 64 + k];
    else if (k < 128) v = hl[(size_t)ns * 64 + (k - 64)];
    else if (k < 192) v = p1[(size_t)n * 64 + (k - 128)];
    else              v = 2.0f * p2[(size_t)n * 64 + (k - 192)] - hl[(size_t)ns * 64 + (k - 192)];
    Xh[i] = (_Float16)v;
}

// ---------------- fused GEMM (WMMA f16->f32) + LSTM gates -------------------
// Block: 256 threads = 8 waves; 32 rows x 256 cols per block.
// Wave w: rows (w>>2)*16, columns (w&3)*16 .. +15 of EVERY gate -> i/f/t/o
// fragments share lane layout, gate math done in registers.
// X tile staged via GLOBAL_LOAD_ASYNC_TO_LDS_B128 (ASYNCcnt path).
#define XS_STRIDE 264   // 256 + 8 halfs pad: kills 16-way LDS bank conflicts

__global__ __launch_bounds__(256)
void gemm_gates_kernel(const _Float16* __restrict__ Xh, const _Float16* __restrict__ Wt,
                       const float* __restrict__ bias, const float* __restrict__ wpeep,
                       const float* __restrict__ c_in,
                       float* __restrict__ h_out, float* __restrict__ c_out) {
    __shared__ __align__(16) _Float16 Xs[32 * XS_STRIDE];
    const int tid = threadIdx.x;
    const int rowblk = blockIdx.x * 32;

    // ---- async-stage 32x256 f16 X tile (16 KB) directly into LDS ----------
    // GVS mode: per-lane LDS addr VGPR, per-lane 32-bit global offset VGPR,
    // uniform SGPR-pair base. Tracked with ASYNCcnt.
    {
        const _Float16* gbase = Xh + (size_t)rowblk * 256;   // uniform -> SGPR pair
        #pragma unroll
        for (int it = 0; it < 4; ++it) {
            int chunk = it * 256 + tid;                 // 1024 x 16B chunks
            int r = chunk >> 5;                         // 0..31
            int cpos = (chunk & 31) * 8;                // half offset in row
            unsigned lds_addr = (unsigned)(size_t)&Xs[r * XS_STRIDE + cpos];
            unsigned goff = (unsigned)((r * 256 + cpos) * 2);   // bytes
            asm volatile("global_load_async_to_lds_b128 %0, %1, %2"
                         :
                         : "v"(lds_addr), "v"(goff), "s"(gbase)
                         : "memory");
        }
        asm volatile("s_wait_asynccnt 0" ::: "memory");
    }
    __syncthreads();

    const int wv = tid >> 5;
    const int lane = tid & 31;
    const int rowTile = (wv >> 2) * 16;       // 0 or 16
    const int colbase = (wv & 3) * 16;        // 0/16/32/48
    const int m = lane & 15;
    const int khalf = lane >> 4;              // 0: K 0..15, 1: K 16..31

    const _Float16* arow = &Xs[(rowTile + m) * XS_STRIDE];
    const int colL = colbase + (lane & 15);

    const _Float16* wcolI = &Wt[(size_t)(0 * 64 + colL) * 256];
    const _Float16* wcolF = &Wt[(size_t)(1 * 64 + colL) * 256];
    const _Float16* wcolT = &Wt[(size_t)(2 * 64 + colL) * 256];
    const _Float16* wcolO = &Wt[(size_t)(3 * 64 + colL) * 256];
    __builtin_prefetch(wcolI, 0, 2);
    __builtin_prefetch(wcolF, 0, 2);
    __builtin_prefetch(wcolT, 0, 2);
    __builtin_prefetch(wcolO, 0, 2);

    v8f accI = {0.f,0.f,0.f,0.f,0.f,0.f,0.f,0.f};
    v8f accF = {0.f,0.f,0.f,0.f,0.f,0.f,0.f,0.f};
    v8f accT = {0.f,0.f,0.f,0.f,0.f,0.f,0.f,0.f};
    v8f accO = {0.f,0.f,0.f,0.f,0.f,0.f,0.f,0.f};

    #pragma unroll
    for (int kk = 0; kk < 8; ++kk) {
        const int k0 = kk * 32;
        // A fragment: lane m holds row m, K = k0 + khalf*16 + {0..15}
        v8h alo = *(const v8h*)&arow[k0 + khalf * 16];
        v8h ahi = *(const v8h*)&arow[k0 + khalf * 16 + 8];
        v16h a = __builtin_shufflevector(alo, ahi, 0,1,2,3,4,5,6,7,8,9,10,11,12,13,14,15);

        const int kb = k0 + khalf * 16;       // B: lanes 0-15 -> K k0..k0+15, 16-31 -> +16
        const _Float16* wpI = wcolI + kb;
        const _Float16* wpF = wcolF + kb;
        const _Float16* wpT = wcolT + kb;
        const _Float16* wpO = wcolO + kb;

        v16h bI = __builtin_shufflevector(*(const v8h*)wpI, *(const v8h*)(wpI + 8),
                                          0,1,2,3,4,5,6,7,8,9,10,11,12,13,14,15);
        v16h bF = __builtin_shufflevector(*(const v8h*)wpF, *(const v8h*)(wpF + 8),
                                          0,1,2,3,4,5,6,7,8,9,10,11,12,13,14,15);
        v16h bT = __builtin_shufflevector(*(const v8h*)wpT, *(const v8h*)(wpT + 8),
                                          0,1,2,3,4,5,6,7,8,9,10,11,12,13,14,15);
        v16h bO = __builtin_shufflevector(*(const v8h*)wpO, *(const v8h*)(wpO + 8),
                                          0,1,2,3,4,5,6,7,8,9,10,11,12,13,14,15);

        accI = __builtin_amdgcn_wmma_f32_16x16x32_f16(false, a, false, bI, (short)0, accI, false, false);
        accF = __builtin_amdgcn_wmma_f32_16x16x32_f16(false, a, false, bF, (short)0, accF, false, false);
        accT = __builtin_amdgcn_wmma_f32_16x16x32_f16(false, a, false, bT, (short)0, accT, false, false);
        accO = __builtin_amdgcn_wmma_f32_16x16x32_f16(false, a, false, bO, (short)0, accO, false, false);
    }

    // ---- LSTM gate epilogue, all in registers -----------------------------
    const float wp0 = wpeep[colL], wp1 = wpeep[64 + colL], wp2 = wpeep[128 + colL];
    const float bI = bias[colL], bF = bias[64 + colL];
    const float bT = bias[128 + colL], bO = bias[192 + colL];
    const int rbase = rowblk + rowTile + (khalf ? 8 : 0);   // C/D: lanes 16-31 -> M = 8+r

    #pragma unroll
    for (int r = 0; r < 8; ++r) {
        int rowG = rbase + r;
        int rc = rowG < N_NODES ? rowG : N_NODES - 1;
        float cl = c_in[(size_t)rc * HID + colL];
        float gi = sigf(accI[r] + bI + wp0 * cl);
        float gf = sigf(accF[r] + bF + wp1 * cl);
        float gt = tanhf(accT[r] + bT);
        float ct = gf * cl + gi * gt;
        float go = sigf(accO[r] + bO + wp2 * ct);
        float ht = go * tanhf(ct);
        if (rowG < N_NODES) {
            h_out[(size_t)rowG * HID + colL] = ht;
            c_out[(size_t)rowG * HID + colL] = ct;
        }
    }
}

// ---------------- final projection: out = h1 @ fc_w.T + fc_b ----------------
// One half-wave (16 lanes) per node: coalesced float4 row read + shfl reduce.
__global__ void fc_kernel(const float* __restrict__ h1, const float* __restrict__ fcw,
                          const float* __restrict__ fcb, float* __restrict__ out) {
    int tid = threadIdx.x;
    int gid = blockIdx.x * blockDim.x + tid;
    int n = gid >> 4;                 // node per 16 lanes
    int q = (gid & 15) * 4;           // 4 floats per lane
    if (n >= N_NODES) return;
    const float4 hv = *(const float4*)(h1 + (size_t)n * 64 + q);
    const float4 wv = *(const float4*)(fcw + q);
    float part = hv.x * wv.x + hv.y * wv.y + hv.z * wv.z + hv.w * wv.w;
    part += __shfl_xor(part, 1, 32);
    part += __shfl_xor(part, 2, 32);
    part += __shfl_xor(part, 4, 32);
    part += __shfl_xor(part, 8, 32);
    if ((gid & 15) == 0) out[n] = part + fcb[0];
}

// ---------------------------------------------------------------------------
extern "C" void kernel_launch(void* const* d_in, const int* in_sizes, int n_in,
                              void* d_out, int out_size, void* d_ws, size_t ws_size,
                              hipStream_t stream) {
    const float* x   = (const float*)d_in[0];
    const int*   ei  = (const int*)d_in[1];
    const float* ew  = (const float*)d_in[2];
    const float* h   = (const float*)d_in[3];
    const float* c   = (const float*)d_in[4];
    const float* Wx  = (const float*)d_in[5];
    const float* Wc  = (const float*)d_in[6];
    const float* bc  = (const float*)d_in[7];
    const float* wp  = (const float*)d_in[8];
    const float* bg  = (const float*)d_in[9];
    const float* fw  = (const float*)d_in[10];
    const float* fb  = (const float*)d_in[11];
    float* out = (float*)d_out;

    const int* row = ei;
    const int* col = ei + E_EDGES;

    // workspace carve-out (256B aligned)
    char* ws = (char*)d_ws;
    size_t off = 0;
    auto carve = [&](size_t bytes) -> void* {
        void* p = ws + off;
        off += (bytes + 255) & ~(size_t)255;
        return p;
    };
    float*    deg  = (float*)carve((size_t)N_PAD * 4);
    float*    nw   = (float*)carve((size_t)E_EDGES * 4);
    float*    p1   = (float*)carve((size_t)N_PAD * HID * 4);
    float*    p2   = (float*)carve((size_t)N_PAD * HID * 4);
    _Float16* Xh   = (_Float16*)carve((size_t)N_PAD * 256 * 2);
    _Float16* Wt   = (_Float16*)carve((size_t)256 * 256 * 2);
    float*    bias = (float*)carve(256 * 4);

    float* h_out = out + N_NODES;                               // (L,N,H)
    float* c_out = out + N_NODES + (size_t)LAY * N_NODES * HID; // (L,N,H)

    const int TB = 256;
    // edge-weight normalization
    zero_f32<<<(N_PAD / 4 + TB - 1) / TB, TB, 0, stream>>>(deg, N_PAD / 4);
    deg_kernel<<<(E_EDGES + TB - 1) / TB, TB, 0, stream>>>(row, ew, deg);
    nw_kernel<<<(E_EDGES + TB - 1) / TB, TB, 0, stream>>>(row, col, ew, deg, nw);

    const float* inp = x;
    const int pn4 = N_PAD * HID / 4;
    for (int l = 0; l < LAY; ++l) {
        const float* hl = h + (size_t)l * N_NODES * HID;
        const float* cl = c + (size_t)l * N_NODES * HID;
        float* hOutL = h_out + (size_t)l * N_NODES * HID;
        float* cOutL = c_out + (size_t)l * N_NODES * HID;

        zero_f32<<<(pn4 + TB - 1) / TB, TB, 0, stream>>>(p1, pn4);
        prop_kernel<<<(E_EDGES * 16 + TB - 1) / TB, TB, 0, stream>>>(row, col, nw, hl, p1);
        zero_f32<<<(pn4 + TB - 1) / TB, TB, 0, stream>>>(p2, pn4);
        prop_kernel<<<(E_EDGES * 16 + TB - 1) / TB, TB, 0, stream>>>(row, col, nw, p1, p2);

        wprep_kernel<<<(256 * 256 + TB - 1) / TB, TB, 0, stream>>>(Wx, Wc, bc, bg, l, Wt, bias);
        xpack_kernel<<<(N_PAD * 256 + TB - 1) / TB, TB, 0, stream>>>(inp, hl, p1, p2, Xh);

        gemm_gates_kernel<<<N_PAD / 32, TB, 0, stream>>>(Xh, Wt, bias,
                                                         wp + (size_t)l * 3 * HID,
                                                         cl, hOutL, cOutL);
        inp = hOutL;
    }

    fc_kernel<<<((N_NODES * 16) + TB - 1) / TB, TB, 0, stream>>>(inp, fw, fb, out);
}
